// LSTMFromScratchForecaster_5171140624836
// MI455X (gfx1250) — matrix operand order repaired and checked by reading, not compiled, not run on hardware
//
#include <hip/hip_runtime.h>
#include <hip/hip_bf16.h>

#define BATCH 128
#define TSEQ  128
#define INDIM 8
#define HDIM  1024
#define NDIM  4096   // 4 gates * H
#define OUTD  2
#define NB    64     // N columns per block
#define NSUB  (NB / 16)
#define KCH   32     // K per chunk (one wmma depth)

typedef __attribute__((ext_vector_type(16))) __bf16 v16bf;
typedef __attribute__((ext_vector_type(8)))  float  v8f;

union Frag { v16bf v; uint4 q[2]; };

__device__ __forceinline__ unsigned short f2bf(float f) {
    unsigned int u = __builtin_bit_cast(unsigned int, f);
    u += 0x7FFFu + ((u >> 16) & 1u);   // round-to-nearest-even
    return (unsigned short)(u >> 16);
}

// ---------------- fp32 -> bf16 weight conversion (grid-stride) ----------------
__global__ void cvt_bf16_kernel(const float* __restrict__ src,
                                unsigned short* __restrict__ dst, int n) {
    for (int i = blockIdx.x * blockDim.x + threadIdx.x; i < n;
         i += gridDim.x * blockDim.x)
        dst[i] = f2bf(src[i]);
}

// ---------------- zero h (bf16) and c (fp32) state ----------------
__global__ void init_state_kernel(unsigned short* __restrict__ hbf,
                                  float* __restrict__ c) {
    int i = blockIdx.x * blockDim.x + threadIdx.x;   // B*H threads
    hbf[i] = 0;
    c[i]   = 0.0f;
}

// Async-stage one 16B slice of the 64x32 bf16 weight chunk into LDS.
__device__ __forceinline__ void stage_async(const unsigned short* __restrict__ Bsrc,
                                            int n0, int k0, unsigned ldsDstByte,
                                            int sr, int sp) {
    // row n0+sr (sr in 0..63), elements k0 + sp*8 .. +7  (16 bytes)
    unsigned long long ga =
        (unsigned long long)(Bsrc + (size_t)(n0 + sr) * HDIM + k0 + sp * 8);
    unsigned la = ldsDstByte + (unsigned)(sr * 64 + sp * 16);
    asm volatile("global_load_async_to_lds_b128 %0, %1, off"
                 :: "v"(la), "v"(ga) : "memory");
}

// ---------------- per-timestep gate GEMM ----------------
// z[m,n] = bias[n] + A1[m,:]·B1[n,:] (+ A2[m,:]·B2[n,:]) (+ x[m,t,:]·Uin[n,:])
// Block: 256 threads = 8 waves; computes a 128(M) x 64(N) strip of z.
// Weight chunks (64 rows x 32 K bf16 = 4KB) are async-staged into
// double-buffered LDS and shared by all 8 waves (each weight row is read
// from L2 exactly once per timestep device-wide).
__global__ __launch_bounds__(256) void lstm_gemm_kernel(
    const unsigned short* __restrict__ A1, int lda1,
    const unsigned short* __restrict__ B1,
    const unsigned short* __restrict__ A2, int lda2,
    const unsigned short* __restrict__ B2,
    const float* __restrict__ bias,   // [4096]
    const float* __restrict__ x,      // [B,T,8] or nullptr
    const float* __restrict__ Uin,    // [4096,8] (layer 0 only)
    int t,
    float* __restrict__ z)            // [B,4096]
{
    __shared__ unsigned short smem[2 * NB * KCH];   // 2 x 4KB

    const int tid  = threadIdx.x;
    const int lane = tid & 31;
    const int wave = tid >> 5;     // 0..7 -> M tile
    const int half = lane >> 4;    // 0/1
    const int lr   = lane & 15;
    const int m0   = wave * 16;
    const int n0   = blockIdx.x * NB;

    // cooperative staging decomposition: 256 threads x 16B = 4KB chunk
    const int sr = tid >> 2;       // 0..63: weight row within strip
    const int sp = tid & 3;        // 0..3 : 16B part of the 64B row-chunk

    const unsigned ldsBase = (unsigned)(size_t)&smem[0];   // low 32b = LDS offset

    v8f acc[NSUB];
#pragma unroll
    for (int j = 0; j < NSUB; ++j) acc[j] = (v8f){};

#pragma unroll 1
    for (int pass = 0; pass < 2; ++pass) {
        const unsigned short* A  = (pass == 0) ? A1 : A2;
        const unsigned short* Bm = (pass == 0) ? B1 : B2;
        if (A == nullptr) break;                     // uniform across block
        const int lda = (pass == 0) ? lda1 : lda2;
        const unsigned short* Ap = A + (size_t)(m0 + lr) * (size_t)lda;

        // prologue: stage chunk 0 into buffer 0
        stage_async(Bm, n0, 0, ldsBase, sr, sp);
        asm volatile("s_wait_asynccnt 0x0" ::: "memory");
        __syncthreads();

#pragma unroll 1
        for (int kc = 0; kc < HDIM / KCH; ++kc) {
            const int k   = kc * KCH;
            const int cur = kc & 1;
            if (kc + 1 < HDIM / KCH)                 // stage next chunk (overlap)
                stage_async(Bm, n0, k + KCH,
                            ldsBase + (unsigned)((cur ^ 1) * NB * KCH * 2), sr, sp);

            // A 16x32 bf16 fragment (ISA 7.12.2 K-interleave), reused 4x
            Frag a;
            a.q[0] = *(const uint4*)(Ap + k + half * 8);
            a.q[1] = *(const uint4*)(Ap + k + 16 + half * 8);

            const unsigned short* sb = &smem[cur * NB * KCH];
#pragma unroll
            for (int j = 0; j < NSUB; ++j) {
                // B 32x16 fragment from LDS: lane = N column, K-interleaved like A
                Frag b;
                const unsigned short* bp = sb + (16 * j + lr) * KCH + half * 8;
                b.q[0] = *(const uint4*)(bp);
                b.q[1] = *(const uint4*)(bp + 16);
                acc[j] = __builtin_amdgcn_wmma_f32_16x16x32_bf16(
                    false, a.v, false, b.v, (short)0, acc[j], false, false);
            }
            asm volatile("s_wait_asynccnt 0x0" ::: "memory");
            __syncthreads();
        }
    }

    // ---- epilogue: bias + (layer 0) K=8 fp32 input projection ----
#pragma unroll
    for (int j = 0; j < NSUB; ++j) {
        const int   n  = n0 + 16 * j + lr;
        const float bn = bias[n];
#pragma unroll
        for (int r = 0; r < 8; ++r) {
            const int m = m0 + r + 8 * half;     // C/D layout: M = r + 8*(lane/16)
            float v = acc[j][r] + bn;
            if (x != nullptr) {
                const float* xr = x + ((size_t)m * TSEQ + t) * INDIM;
                const float* ur = Uin + (size_t)n * INDIM;
                float s = 0.0f;
#pragma unroll
                for (int d = 0; d < INDIM; ++d) s += xr[d] * ur[d];
                v += s;
            }
            z[(size_t)m * NDIM + n] = v;
        }
    }
}

// ---------------- LSTM cell elementwise update ----------------
__global__ __launch_bounds__(256) void lstm_cell_kernel(
    const float* __restrict__ z,          // [B,4096]
    float* __restrict__ c,                // [B,1024]
    unsigned short* __restrict__ hbf,     // [B,1024] bf16 (next GEMM A)
    float* __restrict__ hf,               // [B,1024] fp32 (head input)
    unsigned short* __restrict__ hs_out,  // [B,T,1024] bf16 or nullptr
    int t)
{
    const int i = blockIdx.x * blockDim.x + threadIdx.x;   // B*H threads
    const int b = i >> 10;
    const int k = i & (HDIM - 1);
    const float* zb = z + (size_t)b * NDIM;
    const float zf = zb[k];
    const float zi = zb[HDIM + k];
    const float zo = zb[2 * HDIM + k];
    const float zg = zb[3 * HDIM + k];
    const float f  = 1.0f / (1.0f + expf(-zf));
    const float in = 1.0f / (1.0f + expf(-zi));
    const float o  = 1.0f / (1.0f + expf(-zo));
    const float g  = tanhf(zg);
    const float cn = f * c[i] + in * g;
    c[i] = cn;
    const float hn = o * tanhf(cn);
    hf[i]  = hn;
    hbf[i] = f2bf(hn);
    if (hs_out != nullptr)
        hs_out[((size_t)b * TSEQ + t) * HDIM + k] = f2bf(hn);
}

// ---------------- head: [B,1024] @ [1024,2]^T + b ----------------
__global__ void head_kernel(const float* __restrict__ hf,
                            const float* __restrict__ w,   // [2,1024]
                            const float* __restrict__ bb,  // [2]
                            float* __restrict__ out)       // [B,2]
{
    const int i = blockIdx.x * blockDim.x + threadIdx.x;
    if (i >= BATCH * OUTD) return;
    const int b = i / OUTD, o = i % OUTD;
    const float* hr = hf + (size_t)b * HDIM;
    const float* wr = w + (size_t)o * HDIM;
    float s = 0.0f;
    for (int k = 0; k < HDIM; ++k) s += hr[k] * wr[k];
    out[i] = s + bb[o];
}

extern "C" void kernel_launch(void* const* d_in, const int* in_sizes, int n_in,
                              void* d_out, int out_size, void* d_ws, size_t ws_size,
                              hipStream_t stream) {
    const float* x  = (const float*)d_in[0];   // [B,T,8]
    const float* U0 = (const float*)d_in[1];   // [4,H,8]  -> [4096,8]
    const float* b0 = (const float*)d_in[2];   // [4,H]    -> [4096]
    const float* W0 = (const float*)d_in[3];   // [4,H,H]  -> [4096,1024]
    const float* U1 = (const float*)d_in[4];   // [4,H,H]
    const float* b1 = (const float*)d_in[5];   // [4,H]
    const float* W1 = (const float*)d_in[6];   // [4,H,H]
    const float* hw = (const float*)d_in[7];   // [2,H]
    const float* hb = (const float*)d_in[8];   // [2]
    float* out = (float*)d_out;
    (void)in_sizes; (void)n_in; (void)out_size; (void)ws_size;

    // ---- workspace carve-out (≈60 MB) ----
    char* ws = (char*)d_ws;
    size_t off = 0;
    auto carve = [&](size_t bytes) -> void* {
        void* p = ws + off;
        off = (off + bytes + 255) & ~(size_t)255;
        return p;
    };
    const size_t nW = (size_t)NDIM * HDIM;                       // 4M elements
    unsigned short* W0b  = (unsigned short*)carve(nW * 2);
    unsigned short* U1b  = (unsigned short*)carve(nW * 2);
    unsigned short* W1b  = (unsigned short*)carve(nW * 2);
    unsigned short* hs0b = (unsigned short*)carve((size_t)BATCH * TSEQ * HDIM * 2);
    float*          z    = (float*)carve((size_t)BATCH * NDIM * 4);
    unsigned short* hbf  = (unsigned short*)carve((size_t)BATCH * HDIM * 2);
    float*          c    = (float*)carve((size_t)BATCH * HDIM * 4);
    float*          hf   = (float*)carve((size_t)BATCH * HDIM * 4);

    // ---- one-time per-call weight conversion to bf16 (stays L2-resident) ----
    cvt_bf16_kernel<<<1024, 256, 0, stream>>>(W0, W0b, (int)nW);
    cvt_bf16_kernel<<<1024, 256, 0, stream>>>(U1, U1b, (int)nW);
    cvt_bf16_kernel<<<1024, 256, 0, stream>>>(W1, W1b, (int)nW);

    const int stateBlocks = (BATCH * HDIM) / 256;   // 512
    const int gemmBlocks  = NDIM / NB;              // 64

    // ---- layer 0 ----
    init_state_kernel<<<stateBlocks, 256, 0, stream>>>(hbf, c);
    for (int t = 0; t < TSEQ; ++t) {
        lstm_gemm_kernel<<<gemmBlocks, 256, 0, stream>>>(
            hbf, HDIM, W0b, nullptr, 0, nullptr, b0, x, U0, t, z);
        lstm_cell_kernel<<<stateBlocks, 256, 0, stream>>>(z, c, hbf, hf, hs0b, t);
    }

    // ---- layer 1 (recurrent W1 pass + fused hs0_t @ U1^T pass) ----
    init_state_kernel<<<stateBlocks, 256, 0, stream>>>(hbf, c);
    for (int t = 0; t < TSEQ; ++t) {
        lstm_gemm_kernel<<<gemmBlocks, 256, 0, stream>>>(
            hbf, HDIM, W1b,
            hs0b + (size_t)t * HDIM, TSEQ * HDIM, U1b,
            b1, nullptr, nullptr, t, z);
        lstm_cell_kernel<<<stateBlocks, 256, 0, stream>>>(z, c, hbf, hf, nullptr, t);
    }

    // ---- head ----
    head_kernel<<<1, 256, 0, stream>>>(hf, hw, hb, out);
}
